// TemperGraph_49082886259344
// MI455X (gfx1250) — matrix operand order repaired and checked by reading, not compiled, not run on hardware
//
#include <hip/hip_runtime.h>
#include <hip/hip_bf16.h>

// ---------------------------------------------------------------------------
// TemperGraph on MI455X (gfx1250): bf16 WMMA GEMMs (M=64 tiles), swizzled
// weights, double-buffered async global->LDS staging, WMMA routing MLP.
// ---------------------------------------------------------------------------

typedef __attribute__((ext_vector_type(16))) __bf16 v16bf;
typedef __attribute__((ext_vector_type(8)))  float  v8f;

#define HID     512
#define BSZ     2048
#define INDIM   8192
#define NPATCH  32768      // BSZ * (INDIM/HID)
#define K1VALID 772        // HID + 4 + 256
#define K1PAD   800        // 25 * 32
#define KRVALID 516        // HID + 4
#define KRPAD   544        // 17 * 32
#define NT      64

// ---------------- scalar helpers ----------------
__device__ __forceinline__ unsigned short f2bfbits(float f) {
  unsigned int u; __builtin_memcpy(&u, &f, 4);
  unsigned int r = u + 0x7FFFu + ((u >> 16) & 1u);
  return (unsigned short)(r >> 16);
}

__device__ __forceinline__ unsigned int rotl(unsigned int x, int r) {
  return (x << r) | (x >> (32 - r));
}
// Threefry-2x32-20 (JAX PRNG family), folded output.
__device__ unsigned int threefry(unsigned int k0, unsigned int k1,
                                 unsigned int c0, unsigned int c1) {
  const int R[8] = {13, 15, 26, 6, 17, 29, 16, 24};
  unsigned int ks[3] = {k0, k1, 0x1BD11BDAu ^ k0 ^ k1};
  unsigned int x0 = c0 + k0, x1 = c1 + k1;
#pragma unroll
  for (int b = 0; b < 5; ++b) {
#pragma unroll
    for (int r = 0; r < 4; ++r) {
      x0 += x1;
      x1 = rotl(x1, R[(b & 1) * 4 + r]);
      x1 ^= x0;
    }
    x0 += ks[(b + 1) % 3];
    x1 += ks[(b + 2) % 3] + (unsigned int)(b + 1);
  }
  return x0 ^ x1;
}
__device__ __forceinline__ float gumbel(unsigned int r) {
  float u = ((float)(r >> 9) + 0.5f) * 1.1920929e-7f;
  return -logf(-logf(u));
}

// ---------------- CDNA5 async global->LDS copies ----------------
__device__ __forceinline__ void async_copy16(unsigned lds_byte_addr, const void* gptr) {
  asm volatile("global_load_async_to_lds_b128 %0, %1, off"
               :: "v"(lds_byte_addr), "v"(gptr) : "memory");
}
__device__ __forceinline__ void async_copy8(unsigned lds_byte_addr, const void* gptr) {
  asm volatile("global_load_async_to_lds_b64 %0, %1, off"
               :: "v"(lds_byte_addr), "v"(gptr) : "memory");
}
__device__ __forceinline__ void async_wait0() {
  asm volatile("s_wait_asynccnt 0" ::: "memory");
}
__device__ __forceinline__ void async_wait4() {
  asm volatile("s_wait_asynccnt 4" ::: "memory");
}
__device__ __forceinline__ unsigned lds_addr_of(const void* p) {
  // Shared aperture occupies bits [63:32] of generic LDS addresses; the low
  // 32 bits are the in-LDS byte offset (ISA 10.2).
  return (unsigned)(unsigned long long)p;
}

// ---------------- WMMA fragment loaders (128-bit LDS/global reads) ----------------
// A 16x32 bf16: m = lane&15, half = lane>>4; element j -> K = j + 8*half + 8*(j>=8)
__device__ __forceinline__ v16bf load_a_frag(const unsigned short* base, int stride,
                                             int lane, int kb) {
  const int m = lane & 15, half = lane >> 4;
  union { uint4 q[2]; v16bf v; } u;
  u.q[0] = *(const uint4*)(base + m * stride + kb + 8 * half);
  u.q[1] = *(const uint4*)(base + m * stride + kb + 16 + 8 * half);
  return u.v;
}
// B 32x16 bf16 from swizzled [n][kk]: n = ncol0 + lane&15; element j -> K = j + 8*half + 8*(j>=8)
__device__ __forceinline__ v16bf load_b_frag(const unsigned short* buf, int lane, int ncol0) {
  const int n = ncol0 + (lane & 15), half = lane >> 4;
  const uint4* p = (const uint4*)(buf + n * 32 + 8 * half);
  union { uint4 q[2]; v16bf v; } u;
  u.q[0] = p[0];
  u.q[1] = p[2];
  return u.v;
}
__device__ __forceinline__ v8f wmma_bf16(v16bf a, v16bf b, v8f c) {
  return __builtin_amdgcn_wmma_f32_16x16x32_bf16(false, a, false, b,
                                                 (short)0, c, false, false);
}

// ---------------- init / conversion kernels ----------------
__global__ void k_init_states(const float* __restrict__ x, float* __restrict__ states,
                              unsigned short* __restrict__ states_bf, int n) {
  int i = blockIdx.x * blockDim.x + threadIdx.x;
  if (i < n) {
    float v = x[i];
    states[i] = v;
    states_bf[i] = f2bfbits(v);
  }
}

__global__ void k_init_meta(int* __restrict__ tempers, int* __restrict__ done,
                            const int* __restrict__ seedp) {
  int i = blockIdx.x * blockDim.x + threadIdx.x;
  if (i < NPATCH) {
    unsigned int r = threefry((unsigned int)(*seedp), 0x74656d70u, (unsigned int)i, 0u);
    tempers[i] = (int)(r & (NT - 1));
    done[i] = 0;
  }
}

// f32 [k][ncols_src] -> bf16 swizzled [ks][n<ncols_dst][kk], zero padding.
__global__ void k_cvt_swz(const float* __restrict__ src, unsigned short* __restrict__ dst,
                          int kvalid, int ktot, int ncols_dst, int ncols_src) {
  int i = blockIdx.x * blockDim.x + threadIdx.x;
  if (i < ktot * ncols_dst) {
    int kk = i & 31;
    int n  = (i >> 5) % ncols_dst;
    int ks = i / (ncols_dst * 32);
    int k  = ks * 32 + kk;
    dst[i] = (k < kvalid && n < ncols_src) ? f2bfbits(src[k * ncols_src + n])
                                           : (unsigned short)0;
  }
}

// ---------------- fused hop kernel ----------------
// 512 threads (16 waves) per 64 patch rows. Wave w: row tiles (w>>3)*2+{0,1},
// col tiles (w&7)*4+{0..3}  => acc[2][4] 16x16 tiles of the 64x512 output.
__global__ __launch_bounds__(512) void k_hop(
    const float* __restrict__ op_emb, const float* __restrict__ op_logits,
    const float* __restrict__ id_embeds,
    const float* __restrict__ b1, const float* __restrict__ b2,
    const float* __restrict__ br1, const float* __restrict__ br2,
    const float* __restrict__ tid_emb,
    const unsigned short* __restrict__ W1s, const unsigned short* __restrict__ W2s,
    const unsigned short* __restrict__ Wr1s, const unsigned short* __restrict__ Wr2s,
    float* __restrict__ states, unsigned short* __restrict__ states_bf,
    int* __restrict__ tempers, int* __restrict__ done,
    const int* __restrict__ seedp, int hop) {
  __shared__ __align__(16) unsigned short sB[2][512 * 32]; // 2 x 32 KB weight slices
  __shared__ __align__(16) unsigned short sA[2][64 * 32];  // 2 x 4 KB activation slices
  __shared__ __align__(16) unsigned short sH[64 * HID];    // 64 KB hidden (h1 then h2)
  __shared__ __align__(16) unsigned short sR1[64 * 32];    // routing hidden (bf16)
  __shared__ float sLog[64 * 65];
  __shared__ int sOp[64], sT[64], sDone[64];

  const int tid = threadIdx.x;
  const int lane = tid & 31;
  const int wave = tid >> 5;
  const int row0 = blockIdx.x * 64;
  const int seed = *seedp;

  const int rt0 = (wave >> 3) * 2;   // row tile base (x16 rows)
  const int ct0 = (wave & 7) * 4;    // col tile base (x16 cols)
  const int half = lane >> 4, n0 = lane & 15;

  // ---- async staging helpers ----
  auto issue_sB = [&](const unsigned short* gsrc, int buf) {
#pragma unroll
    for (int c = 0; c < 4; ++c) {
      int e = (c * 512 + tid) * 8;   // 8 ushorts = 16B
      async_copy16(lds_addr_of(&sB[buf][e]), gsrc + e);
    }
  };
  auto issue_sA_x = [&](int ks, int buf) {   // x-portion of x_aug (k < 512)
    int e = tid * 4;                 // 4 ushorts = 8B; covers 64*32 = 2048
    int r = e >> 5, kk = e & 31;
    const unsigned short* gs = states_bf + (size_t)(row0 + r) * HID + ks * 32 + kk;
    async_copy8(lds_addr_of(&sA[buf][e]), gs);
  };

  // prologue: stage slice 0 (GEMM1) before anything else
  issue_sB(W1s, 0);
  issue_sA_x(0, 0);

  if (tid < 64) {
    int row = row0 + tid;
    sT[tid] = tempers[row];
    sDone[tid] = done[row];
    float best = -1e30f; int bi = 0;
    for (int o = 0; o < 3; ++o) {
      float v = op_logits[o] +
                gumbel(threefry((unsigned int)seed, 0x6f703030u + (unsigned int)hop,
                                (unsigned int)row, (unsigned int)o));
      if (v > best) { best = v; bi = o; }
    }
    sOp[tid] = bi;
  }
  __syncthreads();

  v8f acc[2][4];
#pragma unroll
  for (int rr = 0; rr < 2; ++rr)
#pragma unroll
    for (int c = 0; c < 4; ++c)
#pragma unroll
      for (int j = 0; j < 8; ++j) acc[rr][c][j] = 0.f;

  auto mm_step = [&](const unsigned short* aBase, int aStride, int aKb,
                     const unsigned short* bBuf) {
#pragma unroll
    for (int rr = 0; rr < 2; ++rr) {
      v16bf a = load_a_frag(aBase + (rt0 + rr) * 16 * aStride, aStride, lane, aKb);
#pragma unroll
      for (int c = 0; c < 4; ++c) {
        v16bf b = load_b_frag(bBuf, lane, (ct0 + c) * 16);
        acc[rr][c] = wmma_bf16(a, b, acc[rr][c]);
      }
    }
  };

  // ---- GEMM1: x_aug[64 x 800] @ W1[800 x 512], double-buffered ----
  for (int ks = 0; ks < K1PAD / 32; ++ks) {
    const int buf = ks & 1;
    if (ks + 1 < K1PAD / 32) {
      issue_sB(W1s + (size_t)(ks + 1) * (512 * 32), buf ^ 1);
      if (ks + 1 < 16) issue_sA_x(ks + 1, buf ^ 1);
    }
    if (ks >= 16) {  // embedding tail of x_aug: VALU build
      const int kg0 = ks * 32;
      for (int i = tid; i < 64 * 32; i += 512) {
        int r = i >> 5, k = kg0 + (i & 31);
        unsigned short bits;
        if (k < KRVALID)      bits = f2bfbits(id_embeds[sT[r] * 4 + (k - HID)]);
        else if (k < K1VALID) bits = f2bfbits(op_emb[sOp[r] * 256 + (k - KRVALID)]);
        else                  bits = 0;
        sA[buf][i] = bits;
      }
    }
    if (ks + 1 < K1PAD / 32) async_wait4(); else async_wait0();
    __syncthreads();
    mm_step(sA[buf], 32, 0, sB[buf]);
    __syncthreads();
  }

  // stage GEMM2 slice 0 while finishing h1
  issue_sB(W2s, 0);
  // h1 = relu(acc + b1) -> sH (bf16), reset accumulators
#pragma unroll
  for (int rr = 0; rr < 2; ++rr)
#pragma unroll
    for (int c = 0; c < 4; ++c) {
      int n = (ct0 + c) * 16 + n0;
#pragma unroll
      for (int j = 0; j < 8; ++j) {
        int lr = (rt0 + rr) * 16 + j + 8 * half;
        float v = acc[rr][c][j] + b1[n];
        v = v > 0.f ? v : 0.f;
        sH[lr * HID + n] = f2bfbits(v);
        acc[rr][c][j] = 0.f;
      }
    }

  // ---- GEMM2: h1[64 x 512] @ W2[512 x 512], double-buffered ----
  for (int ks = 0; ks < HID / 32; ++ks) {
    const int buf = ks & 1;
    if (ks + 1 < HID / 32) issue_sB(W2s + (size_t)(ks + 1) * (512 * 32), buf ^ 1);
    if (ks + 1 < HID / 32) async_wait4(); else async_wait0();
    __syncthreads();
    mm_step(sH, HID, ks * 32, sB[buf]);
    __syncthreads();
  }

  // h2 = relu(acc + b2); commit state for not-done rows; keep bf16 in sH
#pragma unroll
  for (int rr = 0; rr < 2; ++rr)
#pragma unroll
    for (int c = 0; c < 4; ++c) {
      int n = (ct0 + c) * 16 + n0;
#pragma unroll
      for (int j = 0; j < 8; ++j) {
        int lr = (rt0 + rr) * 16 + j + 8 * half;
        float v = acc[rr][c][j] + b2[n];
        v = v > 0.f ? v : 0.f;
        unsigned short bits = f2bfbits(v);
        sH[lr * HID + n] = bits;
        if (!sDone[lr]) {
          states[(size_t)(row0 + lr) * HID + n] = v;
          states_bf[(size_t)(row0 + lr) * HID + n] = bits;
        }
      }
    }
  // tid_emb tail slice of enriched (K = 512..543) into sA[0]
  for (int i = tid; i < 64 * 32; i += 512) {
    int r = i >> 5, k = HID + (i & 31);
    sA[0][i] = (k < KRVALID) ? f2bfbits(tid_emb[sT[r] * 4 + (k - HID)])
                             : (unsigned short)0;
  }
  __syncthreads();

  // ---- routing GEMM A: enriched[64 x 544] @ Wr1[544 x 32] (waves 0..7) ----
  if (wave < 8) {
    const int rt = wave >> 1, ct = wave & 1;
    v8f accR;
#pragma unroll
    for (int j = 0; j < 8; ++j) accR[j] = 0.f;
    for (int ks = 0; ks < KRPAD / 32; ++ks) {
      v16bf a = (ks < 16) ? load_a_frag(sH + rt * 16 * HID, HID, lane, ks * 32)
                          : load_a_frag(sA[0] + rt * 16 * 32, 32, lane, 0);
      v16bf b = load_b_frag(Wr1s + ks * (32 * 32), lane, ct * 16);
      accR = wmma_bf16(a, b, accR);
    }
    int n = ct * 16 + n0;
#pragma unroll
    for (int j = 0; j < 8; ++j) {
      int lr = rt * 16 + j + 8 * half;
      float v = accR[j] + br1[n];
      v = v > 0.f ? v : 0.f;
      sR1[lr * 32 + n] = f2bfbits(v);
    }
  }
  __syncthreads();

  // ---- routing GEMM B: r1[64 x 32] @ Wr2[32 x 80] -> logits[64 x 65] ----
  for (int tt = wave; tt < 20; tt += 16) {
    const int rt = tt / 5, ct = tt % 5;
    v8f acc2;
#pragma unroll
    for (int j = 0; j < 8; ++j) acc2[j] = 0.f;
    v16bf a = load_a_frag(sR1 + rt * 16 * 32, 32, lane, 0);
    v16bf b = load_b_frag(Wr2s, lane, ct * 16);
    acc2 = wmma_bf16(a, b, acc2);
    int n = ct * 16 + n0;
    if (n < 65) {
#pragma unroll
      for (int j = 0; j < 8; ++j) {
        int lr = rt * 16 + j + 8 * half;
        sLog[lr * 65 + n] = acc2[j] + br2[n];
      }
    }
  }
  __syncthreads();

  // categorical(log_softmax(logits)) == Gumbel-argmax over logits
  if (tid < 64 && !sDone[tid]) {
    int row = row0 + tid;
    float best = -1e30f; int bj = 0;
    for (int j = 0; j < 65; ++j) {
      float v = sLog[tid * 65 + j] +
                gumbel(threefry((unsigned int)seed, 0x72743030u + (unsigned int)hop,
                                (unsigned int)row, (unsigned int)j));
      if (v > best) { best = v; bj = j; }
    }
    if (bj == NT) { done[row] = 1; tempers[row] = NT - 1; }
    else          { tempers[row] = bj; }
  }
}

// ---------------- mean over patches ----------------
__global__ void k_mean(const float* __restrict__ states, float* __restrict__ latent_out,
                       unsigned short* __restrict__ latent_bf, int n) {
  int i = blockIdx.x * blockDim.x + threadIdx.x;
  if (i < n) {
    int b = i >> 9, c = i & (HID - 1);
    float s = 0.f;
#pragma unroll
    for (int p = 0; p < 16; ++p) s += states[(size_t)(b * 16 + p) * HID + c];
    s *= (1.0f / 16.0f);
    latent_out[i] = s;
    latent_bf[i] = f2bfbits(s);
  }
}

// ---------------- pred = latent @ Wp + bp (WMMA, M=64) ----------------
__global__ __launch_bounds__(512) void k_pred(
    const unsigned short* __restrict__ latent_bf, const unsigned short* __restrict__ Wps,
    const float* __restrict__ bp, float* __restrict__ pred) {
  __shared__ __align__(16) unsigned short sA2[64 * HID];   // 64 KB latent tile
  __shared__ __align__(16) unsigned short sB[2][512 * 32]; // 2 x 32 KB Wp slices
  const int tid = threadIdx.x, lane = tid & 31, wave = tid >> 5;
  const int row0 = blockIdx.x * 64;
  const int cb0 = blockIdx.y * 512;
  const int rt0 = (wave >> 3) * 2;
  const int ct0 = (wave & 7) * 4;
  const int half = lane >> 4, n0 = lane & 15;

  auto issue_sB = [&](const unsigned short* gsrc, int buf) {
#pragma unroll
    for (int c = 0; c < 4; ++c) {
      int e = (c * 512 + tid) * 8;
      async_copy16(lds_addr_of(&sB[buf][e]), gsrc + e);
    }
  };

  // stage the full 64x512 latent tile (contiguous 64 KB) + Wp slice 0
  {
    const unsigned short* gsrc = latent_bf + (size_t)row0 * HID;
#pragma unroll
    for (int c = 0; c < 8; ++c) {
      int e = (c * 512 + tid) * 8;
      async_copy16(lds_addr_of(sA2 + e), gsrc + e);
    }
  }
  issue_sB(Wps + (size_t)cb0 * 32, 0);

  v8f acc[2][4];
#pragma unroll
  for (int rr = 0; rr < 2; ++rr)
#pragma unroll
    for (int c = 0; c < 4; ++c)
#pragma unroll
      for (int j = 0; j < 8; ++j) acc[rr][c][j] = 0.f;

  for (int ks = 0; ks < HID / 32; ++ks) {
    const int buf = ks & 1;
    if (ks + 1 < HID / 32)
      issue_sB(Wps + ((size_t)(ks + 1) * INDIM + cb0) * 32, buf ^ 1);
    if (ks + 1 < HID / 32) async_wait4(); else async_wait0();
    __syncthreads();
#pragma unroll
    for (int rr = 0; rr < 2; ++rr) {
      v16bf a = load_a_frag(sA2 + (rt0 + rr) * 16 * HID, HID, lane, ks * 32);
#pragma unroll
      for (int c = 0; c < 4; ++c) {
        v16bf b = load_b_frag(sB[buf], lane, (ct0 + c) * 16);
        acc[rr][c] = wmma_bf16(a, b, acc[rr][c]);
      }
    }
    __syncthreads();
  }
#pragma unroll
  for (int rr = 0; rr < 2; ++rr)
#pragma unroll
    for (int c = 0; c < 4; ++c) {
      int n = (ct0 + c) * 16 + n0;
#pragma unroll
      for (int j = 0; j < 8; ++j) {
        int m = (rt0 + rr) * 16 + j + 8 * half;
        pred[(size_t)(row0 + m) * INDIM + cb0 + n] = acc[rr][c][j] + bp[cb0 + n];
      }
    }
}

// ---------------------------------------------------------------------------
extern "C" void kernel_launch(void* const* d_in, const int* in_sizes, int n_in,
                              void* d_out, int out_size, void* d_ws, size_t ws_size,
                              hipStream_t stream) {
  const float* x         = (const float*)d_in[0];
  const float* op_emb    = (const float*)d_in[1];
  const float* op_logits = (const float*)d_in[2];
  const float* id_embeds = (const float*)d_in[3];
  const float* W1        = (const float*)d_in[4];
  const float* b1        = (const float*)d_in[5];
  const float* W2        = (const float*)d_in[6];
  const float* b2        = (const float*)d_in[7];
  const float* Wr1       = (const float*)d_in[8];
  const float* br1       = (const float*)d_in[9];
  const float* Wr2       = (const float*)d_in[10];
  const float* br2       = (const float*)d_in[11];
  const float* Wp        = (const float*)d_in[12];
  const float* bp        = (const float*)d_in[13];
  const float* tid_emb   = (const float*)d_in[14];
  const int*   seedp     = (const int*)d_in[15];

  char* ws = (char*)d_ws;
  size_t off = 0;
  auto carve = [&](size_t bytes) -> char* {
    char* p = ws + off;
    off = (off + bytes + 255) & ~(size_t)255;
    return p;
  };
  float*          states    = (float*)carve((size_t)NPATCH * HID * 4);
  unsigned short* states_bf = (unsigned short*)carve((size_t)NPATCH * HID * 2);
  int*            tempers   = (int*)carve((size_t)NPATCH * 4);
  int*            done      = (int*)carve((size_t)NPATCH * 4);
  unsigned short* W1s       = (unsigned short*)carve((size_t)K1PAD * HID * 2);
  unsigned short* W2s       = (unsigned short*)carve((size_t)HID * HID * 2);
  unsigned short* Wps       = (unsigned short*)carve((size_t)HID * INDIM * 2);
  unsigned short* Wr1s      = (unsigned short*)carve((size_t)KRPAD * 32 * 2);
  unsigned short* Wr2s      = (unsigned short*)carve((size_t)32 * 80 * 2);
  unsigned short* latent_bf = (unsigned short*)carve((size_t)BSZ * HID * 2);
  (void)ws_size; (void)n_in; (void)in_sizes; (void)out_size;

  const int TPB = 256;
  k_init_states<<<(BSZ * INDIM + TPB - 1) / TPB, TPB, 0, stream>>>(x, states, states_bf,
                                                                   BSZ * INDIM);
  k_init_meta<<<(NPATCH + TPB - 1) / TPB, TPB, 0, stream>>>(tempers, done, seedp);
  k_cvt_swz<<<(K1PAD * HID + TPB - 1) / TPB, TPB, 0, stream>>>(W1, W1s, K1VALID, K1PAD,
                                                               HID, HID);
  k_cvt_swz<<<(HID * HID + TPB - 1) / TPB, TPB, 0, stream>>>(W2, W2s, HID, HID, HID, HID);
  k_cvt_swz<<<(HID * INDIM + TPB - 1) / TPB, TPB, 0, stream>>>(Wp, Wps, HID, HID,
                                                               INDIM, INDIM);
  k_cvt_swz<<<(KRPAD * 32 + TPB - 1) / TPB, TPB, 0, stream>>>(Wr1, Wr1s, KRVALID, KRPAD,
                                                              32, 32);
  k_cvt_swz<<<(32 * 80 + TPB - 1) / TPB, TPB, 0, stream>>>(Wr2, Wr2s, 32, 32, 80, 65);

  for (int hop = 0; hop < 4; ++hop) {
    k_hop<<<NPATCH / 64, 512, 0, stream>>>(op_emb, op_logits, id_embeds, b1, b2,
                                           br1, br2, tid_emb,
                                           W1s, W2s, Wr1s, Wr2s,
                                           states, states_bf,
                                           tempers, done, seedp, hop);
  }

  float* out = (float*)d_out;
  k_mean<<<(BSZ * HID + TPB - 1) / TPB, TPB, 0, stream>>>(states, out, latent_bf,
                                                          BSZ * HID);
  dim3 pgrid(BSZ / 64, INDIM / 512);
  k_pred<<<pgrid, 512, 0, stream>>>(latent_bf, Wps, bp, out + (size_t)BSZ * HID);
}
